// Net_8813272891728
// MI455X (gfx1250) — matrix operand order repaired and checked by reading, not compiled, not run on hardware
//
#include <hip/hip_runtime.h>
#include <stdint.h>
#include <stddef.h>

// ---------------------------------------------------------------------------
// Types for WMMA fragments (CDNA5 / gfx1250, wave32)
// ---------------------------------------------------------------------------
typedef __attribute__((ext_vector_type(16))) __bf16 v16bf;
typedef __attribute__((ext_vector_type(8)))  __bf16 v8bf;
typedef __attribute__((ext_vector_type(8)))  float  v8f;

union Frag {
    v16bf v;
    v8bf  h[2];
};

// ---------------------------------------------------------------------------
// Zero-fill (u32 granularity, grid-stride)
// ---------------------------------------------------------------------------
__global__ void zero_kernel(uint32_t* __restrict__ p, size_t n) {
    size_t i = (size_t)blockIdx.x * blockDim.x + threadIdx.x;
    size_t s = (size_t)gridDim.x * blockDim.x;
    for (; i < n; i += s) p[i] = 0u;
}

// ---------------------------------------------------------------------------
// Scatter sparse points into dense bf16 activation (Ci padded to 32) + mask
// ---------------------------------------------------------------------------
__global__ void scatter_kernel(const float* __restrict__ feats,
                               const int* __restrict__ coors, int N,
                               __bf16* __restrict__ act, float* __restrict__ m0) {
    int p = blockIdx.x * blockDim.x + threadIdx.x;
    if (p >= N) return;
    int z = coors[p * 4 + 1];
    int y = coors[p * 4 + 2];
    int x = coors[p * 4 + 3];
    size_t v = ((size_t)z * 128 + y) * 32 + x;
    m0[v] = 1.0f;
    act[v * 32 + 0] = (__bf16)feats[p * 3 + 0];
    act[v * 32 + 1] = (__bf16)feats[p * 3 + 1];
    act[v * 32 + 2] = (__bf16)feats[p * 3 + 2];
}

// ---------------------------------------------------------------------------
// Repack f32 weights (kz,ky,kx,Ci,Co) into WMMA B-fragment order, bf16.
// Packed block (tap, kc, nt): 32 lanes x 16 contiguous bf16 per lane:
//   value(lane, e) = W[tap][kc*32 + (lane>>4)*8 + (e<8 ? e : 8+e)][nt*16 + (lane&15)]
// ---------------------------------------------------------------------------
__global__ void repack_kernel(const float* __restrict__ w, __bf16* __restrict__ wpk,
                              int Ci, int Co, int Kc, int Nt, long total) {
    long idx = (long)blockIdx.x * blockDim.x + threadIdx.x;
    if (idx >= total) return;
    int  e    = (int)(idx & 15);
    int  lane = (int)((idx >> 4) & 31);
    long blk  = idx >> 9;
    int  nt   = (int)(blk % Nt);
    long b2   = blk / Nt;
    int  kc   = (int)(b2 % Kc);
    int  tap  = (int)(b2 / Kc);
    int  n    = nt * 16 + (lane & 15);
    int  k    = kc * 32 + (lane >> 4) * 8 + (e < 8 ? e : 8 + e);
    float v = 0.0f;
    if (k < Ci && n < Co) v = w[((size_t)tap * Ci + k) * Co + n];
    wpk[idx] = (__bf16)v;
}

// ---------------------------------------------------------------------------
// Mask pooling: reduce_window(max, k,s,pad) over a (D,H,W) float mask
// ---------------------------------------------------------------------------
__global__ void maskpool_kernel(const float* __restrict__ mi, float* __restrict__ mo,
                                int Di, int Hi, int Wi,
                                int Do, int Ho, int Wo,
                                int k, int s, int pad) {
    int idx = blockIdx.x * blockDim.x + threadIdx.x;
    int n = Do * Ho * Wo;
    if (idx >= n) return;
    int z = idx / (Ho * Wo);
    int r = idx % (Ho * Wo);
    int y = r / Wo;
    int x = r % Wo;
    float best = 0.0f;
    for (int dz = 0; dz < k; ++dz) {
        int iz = z * s - pad + dz;
        if (iz < 0 || iz >= Di) continue;
        for (int dy = 0; dy < k; ++dy) {
            int iy = y * s - pad + dy;
            if (iy < 0 || iy >= Hi) continue;
            for (int dx = 0; dx < k; ++dx) {
                int ix = x * s - pad + dx;
                if (ix < 0 || ix >= Wi) continue;
                float m = mi[((size_t)iz * Hi + iy) * Wi + ix];
                best = best > m ? best : m;
            }
        }
    }
    mo[idx] = best;
}

// ---------------------------------------------------------------------------
// Masked 2x2x2 stride-2 max pool (bf16 activations, float mask on input res)
// ---------------------------------------------------------------------------
__global__ void maxpool_kernel(const __bf16* __restrict__ in, const float* __restrict__ m,
                               __bf16* __restrict__ out,
                               int Di, int Hi, int Wi, int C,
                               int Do, int Ho, int Wo) {
    size_t idx = (size_t)blockIdx.x * blockDim.x + threadIdx.x;
    size_t n = (size_t)Do * Ho * Wo * C;
    if (idx >= n) return;
    int c = (int)(idx % C);
    int v = (int)(idx / C);
    int z = v / (Ho * Wo);
    int r = v % (Ho * Wo);
    int y = r / Wo;
    int x = r % Wo;
    float best = -3.4e38f;
    bool any = false;
    for (int dz = 0; dz < 2; ++dz)
        for (int dy = 0; dy < 2; ++dy)
            for (int dx = 0; dx < 2; ++dx) {
                int iz = 2 * z + dz, iy = 2 * y + dy, ix = 2 * x + dx;
                size_t iv = ((size_t)iz * Hi + iy) * Wi + ix;
                if (m[iv] > 0.0f) {
                    float xv = (float)in[iv * C + c];
                    best = best > xv ? best : xv;
                    any = true;
                }
            }
    out[idx] = (__bf16)(any ? best : 0.0f);
}

// ---------------------------------------------------------------------------
// Implicit-GEMM conv / transposed-conv, v_wmma_f32_16x16x32_bf16.
// Block = 4 waves (128 threads): each wave owns a 2x16-voxel M pair, all
// waves share one 32-channel N pair. The 2KB B tile per (tap,kchunk) is
// async-copied into LDS once per block (global_load_async_to_lds_b128,
// double-buffered, s_wait_asynccnt + workgroup barrier), then each wave
// ds_loads its fragments. A fragments are direct global b128 loads.
// All spatial dims are powers of two -> shift/mask decode, no idiv.
//   MODE 0: conv, grid = (ceil(nAvox/128), Co/32, 1); A rows = output voxels
//   MODE 1: transposed conv k2 s2, grid = (ceil(nAvox/128), Co/32, 8);
//           A rows = input voxels, blockIdx.z = output-parity tap
// ---------------------------------------------------------------------------
template <int MODE>
__global__ __launch_bounds__(128) void conv_wmma_kernel(
    const __bf16* __restrict__ in, const __bf16* __restrict__ wpk,
    const float* __restrict__ mask,
    __bf16* __restrict__ outb, float* __restrict__ outf,
    int Di, int logHi, int logWi, int Ci,
    int logHo, int logWo, int Co,
    int nAvox, int logAH, int logAW,
    int ksz, int stride, int pad) {
    __shared__ __align__(16) char smemB[4096];  // 2 stages x (2 N tiles x 1KB)

    const int tid  = threadIdx.x;
    const int lane = tid & 31;
    const int wv   = tid >> 5;
    const int hlf  = lane >> 4;
    const int col  = lane & 15;
    const int mt = (blockIdx.x * 4 + wv) * 2;  // this wave's 2 M tiles
    const int nt = blockIdx.y * 2;             // block-shared 2 N tiles
    const int Nt = Co >> 4;
    const int Kc = Ci >> 5;
    const int Hi = 1 << logHi, Wi = 1 << logWi;
    const int maskAH = (1 << logAH) - 1, maskAW = (1 << logAW) - 1;
    const int kk2 = ksz * ksz;
    const int T = MODE ? 1 : kk2 * ksz;

    const unsigned smbase = (unsigned)(size_t)&smemB[0];
    const __bf16* wsrc0 = wpk + (size_t)nt * 512 + (size_t)tid * 8;  // +16B per tid

    // Issue the 2KB B tile for (tap,kc) into LDS stage stg (1 b128 per thread).
    auto issueB = [&](int tap, int kc, int stg) {
        const __bf16* src = wsrc0 + (size_t)(tap * Kc + kc) * Nt * 512;
        unsigned dst = smbase + (unsigned)(stg * 2048 + tid * 16);
        asm volatile("global_load_async_to_lds_b128 %0, %1, off"
                     :: "v"(dst), "v"((unsigned long long)(size_t)src)
                     : "memory");
    };

    // Per-M-tile A-row voxel decode (shifts only; garbage beyond nAvox gated)
    int az[2], ay[2], ax[2];
    bool aok[2];
#pragma unroll
    for (int m = 0; m < 2; ++m) {
        const int av = (mt + m) * 16 + col;
        aok[m] = av < nAvox;
        az[m] = av >> (logAH + logAW);
        ay[m] = (av >> logAW) & maskAH;
        ax[m] = av & maskAW;
    }

    v8f acc[2][2] = {{{}, {}}, {{}, {}}};

    issueB(MODE ? (int)blockIdx.z : 0, 0, 0);  // prologue: stage 0
    int it = 0;

    for (int t = 0; t < T; ++t) {
        const int tap = MODE ? (int)blockIdx.z : t;
        const __bf16* ib[2];
        bool iok[2];
        if (MODE) {
#pragma unroll
            for (int m = 0; m < 2; ++m) {
                iok[m] = aok[m];
                ib[m] = in + ((size_t)(((az[m] << logHi) + ay[m]) << logWi) + ax[m]) * Ci +
                        hlf * 8;
            }
        } else {
            const int dz = tap / kk2, dy = (tap / ksz) % ksz, dx = tap % ksz;
#pragma unroll
            for (int m = 0; m < 2; ++m) {
                const int iz = az[m] * stride - pad + dz;
                const int iy = ay[m] * stride - pad + dy;
                const int ix = ax[m] * stride - pad + dx;
                iok[m] = aok[m] && (unsigned)iz < (unsigned)Di &&
                         (unsigned)iy < (unsigned)Hi && (unsigned)ix < (unsigned)Wi;
                ib[m] = in + ((size_t)(((iz << logHi) + iy) << logWi) + ix) * Ci + hlf * 8;
            }
        }
        for (int kc = 0; kc < Kc; ++kc, ++it) {
            const int stage = it & 1;
            // Wait own async copy, then barrier: all waves' stage data resident,
            // and the other stage is no longer being read by anyone.
            asm volatile("s_wait_asynccnt 0x0" ::: "memory");
            __syncthreads();
            // Kick off the next B tile into the other stage.
            {
                int nk = kc + 1, ntp = t;
                if (nk == Kc) { nk = 0; ++ntp; }
                if (ntp < T) issueB(MODE ? (int)blockIdx.z : ntp, nk, (it + 1) & 1);
            }
            // A fragments: two guarded 16B global loads per lane per M tile.
            Frag a[2];
#pragma unroll
            for (int m = 0; m < 2; ++m) {
                v8bf a0 = {}, a1 = {};
                if (iok[m]) {
                    const __bf16* p = ib[m] + kc * 32;
                    a0 = *(const v8bf*)(p);
                    a1 = *(const v8bf*)(p + 16);
                }
                a[m].h[0] = a0;
                a[m].h[1] = a1;
            }
            // B fragments from LDS (ds_load_b128 x4).
            const char* bsm = smemB + stage * 2048 + lane * 32;
            Frag b[2];
#pragma unroll
            for (int n = 0; n < 2; ++n) {
                b[n].h[0] = *(const v8bf*)(bsm + n * 1024);
                b[n].h[1] = *(const v8bf*)(bsm + n * 1024 + 16);
            }
#pragma unroll
            for (int m = 0; m < 2; ++m)
#pragma unroll
                for (int n = 0; n < 2; ++n)
                    acc[m][n] = __builtin_amdgcn_wmma_f32_16x16x32_bf16(
                        false, a[m].v, false, b[n].v, (short)0, acc[m][n], false, false);
        }
    }

    // C/D layout: VGPR i, lane l -> row M=(l>>4)*8+i, col N=l&15
#pragma unroll
    for (int m = 0; m < 2; ++m) {
#pragma unroll
        for (int i = 0; i < 8; ++i) {
            const int ar = (mt + m) * 16 + hlf * 8 + i;
            if (ar < nAvox) {
                int ovox;
                if (MODE) {
                    const int z = ar >> (logAH + logAW);
                    const int y = (ar >> logAW) & maskAH;
                    const int x = ar & maskAW;
                    const int tap = (int)blockIdx.z;
                    const int dz = tap >> 2, dy = (tap >> 1) & 1, dx = tap & 1;
                    ovox = (((((2 * z + dz) << logHo) + (2 * y + dy)) << logWo) + (2 * x + dx));
                } else {
                    ovox = ar;
                }
                const float mv = mask[ovox];
#pragma unroll
                for (int n = 0; n < 2; ++n) {
                    const float val = acc[m][n][i] * mv;
                    const size_t off = (size_t)ovox * Co + (nt + n) * 16 + col;
                    if (outf) outf[off] = val;
                    else      outb[off] = (__bf16)val;
                }
            }
        }
    }
}

// ---------------------------------------------------------------------------
// Host-side orchestration
// ---------------------------------------------------------------------------
static inline int ilog2i(int v) {
    int l = 0;
    while ((1 << l) < v) ++l;
    return l;
}

extern "C" void kernel_launch(void* const* d_in, const int* in_sizes, int n_in,
                              void* d_out, int out_size, void* d_ws, size_t ws_size,
                              hipStream_t stream) {
    (void)n_in; (void)out_size; (void)ws_size;
    const float* feats = (const float*)d_in[0];
    const int*   coors = (const int*)d_in[1];
    const int N = in_sizes[0] / 3;
    const float* w[18];
    for (int i = 0; i < 18; ++i) w[i] = (const float*)d_in[3 + i];

    struct LS { int k, ci, co; };
    static const LS L[18] = {
        {3, 3, 32},   {3, 32, 32},  {3, 32, 64},  {3, 64, 64},  {3, 64, 64},
        {3, 64, 96},  {3, 96, 96},  {2, 96, 96},  {3, 96, 128}, {3, 128, 128},
        {3, 128, 160},{3, 160, 160},{3, 160, 192},{3, 192, 192},{3, 192, 224},
        {3, 224, 224},{2, 224, 128},{2, 128, 64}};

    // ---- workspace carve-up (all offsets 256B aligned) ----
    char* base = (char*)d_ws;
    size_t off = 0;
    auto carve = [&](size_t bytes) -> char* {
        char* p = base + off;
        off = (off + bytes + 255) & ~(size_t)255;
        return p;
    };
    __bf16* actA = (__bf16*)carve(67108864);   // 524288 voxels x 64ch bf16 max
    __bf16* actB = (__bf16*)carve(67108864);
    float* m0 = (float*)carve(524288 * 4);
    float* m1 = (float*)carve(65536 * 4);
    float* m2 = (float*)carve(8192 * 4);
    float* m3 = (float*)carve(1024 * 4);
    float* m4 = (float*)carve(128 * 4);
    float* m5 = (float*)carve(16 * 4);
    __bf16* wp[18];
    long wtot[18];
    for (int i = 0; i < 18; ++i) {
        const int T  = L[i].k * L[i].k * L[i].k;
        const int Kc = (L[i].ci + 31) / 32;
        const int Nt = L[i].co / 16;
        wtot[i] = (long)T * Kc * Nt * 512;
        wp[i] = (__bf16*)carve((size_t)wtot[i] * 2);
    }

    // ---- init: zero stage-0 activation (Ci padded 3->32) and m0, scatter ----
    zero_kernel<<<2048, 256, 0, stream>>>((uint32_t*)actA, (size_t)524288 * 32 * 2 / 4);
    zero_kernel<<<2048, 256, 0, stream>>>((uint32_t*)m0, 524288);
    for (int i = 0; i < 18; ++i) {
        const int Kc = (L[i].ci + 31) / 32;
        const int Nt = L[i].co / 16;
        repack_kernel<<<(unsigned)((wtot[i] + 255) / 256), 256, 0, stream>>>(
            w[i], wp[i], L[i].ci, L[i].co, Kc, Nt, wtot[i]);
    }
    scatter_kernel<<<(N + 255) / 256, 256, 0, stream>>>(feats, coors, N, actA, m0);

    auto conv = [&](const __bf16* in, int li, const float* mask, void* out, bool f32out,
                    int Di, int Hi, int Wi, int Ci, int Do, int Ho, int Wo, int Co,
                    int stride, int pad, int mode) {
        const int nAvox = mode ? Di * Hi * Wi : Do * Ho * Wo;
        const int logHi = ilog2i(Hi), logWi = ilog2i(Wi);
        const int logHo = ilog2i(Ho), logWo = ilog2i(Wo);
        const int logAH = mode ? logHi : logHo;
        const int logAW = mode ? logWi : logWo;
        dim3 g((unsigned)((nAvox + 127) / 128), (unsigned)(Co / 32), (unsigned)(mode ? 8 : 1));
        __bf16* ob = f32out ? nullptr : (__bf16*)out;
        float*  of = f32out ? (float*)out : nullptr;
        if (mode)
            conv_wmma_kernel<1><<<g, 128, 0, stream>>>(
                in, wp[li], mask, ob, of, Di, logHi, logWi, Ci, logHo, logWo, Co,
                nAvox, logAH, logAW, L[li].k, stride, pad);
        else
            conv_wmma_kernel<0><<<g, 128, 0, stream>>>(
                in, wp[li], mask, ob, of, Di, logHi, logWi, Ci, logHo, logWo, Co,
                nAvox, logAH, logAW, L[li].k, stride, pad);
    };
    auto mpool = [&](const float* mi, float* mo, int Di, int Hi, int Wi,
                     int Do, int Ho, int Wo, int k, int s, int pad) {
        int n = Do * Ho * Wo;
        maskpool_kernel<<<(n + 255) / 256, 256, 0, stream>>>(mi, mo, Di, Hi, Wi,
                                                             Do, Ho, Wo, k, s, pad);
    };
    auto xpool = [&](const __bf16* in, const float* m, __bf16* out,
                     int Di, int Hi, int Wi, int C, int Do, int Ho, int Wo) {
        size_t n = (size_t)Do * Ho * Wo * C;
        maxpool_kernel<<<(unsigned)((n + 255) / 256), 256, 0, stream>>>(
            in, m, out, Di, Hi, Wi, C, Do, Ho, Wo);
    };

    // ---- stage 0: full resolution (128,128,32), submanifold convs ----
    conv(actA, 0, m0, actB, false, 128,128,32,32, 128,128,32,32, 1,1,0);  // w1
    conv(actB, 1, m0, actA, false, 128,128,32,32, 128,128,32,32, 1,1,0);  // w2
    conv(actA, 2, m0, actB, false, 128,128,32,32, 128,128,32,64, 1,1,0);  // w3
    conv(actB, 3, m0, actA, false, 128,128,32,64, 128,128,32,64, 1,1,0);  // w4
    // ---- downsample to (64,64,16) ----
    mpool(m0, m1, 128,128,32, 64,64,16, 3,2,1);
    conv(actA, 4, m1, actB, false, 128,128,32,64, 64,64,16,64, 2,1,0);    // w5 s2
    conv(actB, 5, m1, actA, false, 64,64,16,64, 64,64,16,96, 1,1,0);      // w6
    conv(actA, 6, m1, actB, false, 64,64,16,96, 64,64,16,96, 1,1,0);      // w7
    // ---- downsample to (32,32,8) ----
    mpool(m1, m2, 64,64,16, 32,32,8, 2,2,0);
    conv(actB, 7, m2, actA, false, 64,64,16,96, 32,32,8,96, 2,0,0);       // w8 k2 s2
    conv(actA, 8, m2, actB, false, 32,32,8,96, 32,32,8,128, 1,1,0);       // w9
    conv(actB, 9, m2, actA, false, 32,32,8,128, 32,32,8,128, 1,1,0);      // w10
    // ---- max pool to (16,16,4) ----
    mpool(m2, m3, 32,32,8, 16,16,4, 2,2,0);
    xpool(actA, m2, actB, 32,32,8,128, 16,16,4);
    conv(actB, 10, m3, actA, false, 16,16,4,128, 16,16,4,160, 1,1,0);     // w11
    conv(actA, 11, m3, actB, false, 16,16,4,160, 16,16,4,160, 1,1,0);     // w12
    // ---- max pool to (8,8,2) ----
    mpool(m3, m4, 16,16,4, 8,8,2, 2,2,0);
    xpool(actB, m3, actA, 16,16,4,160, 8,8,2);
    conv(actA, 12, m4, actB, false, 8,8,2,160, 8,8,2,192, 1,1,0);         // w13
    conv(actB, 13, m4, actA, false, 8,8,2,192, 8,8,2,192, 1,1,0);         // w14
    // ---- max pool to (4,4,1) ----
    mpool(m4, m5, 8,8,2, 4,4,1, 2,2,0);
    xpool(actA, m4, actB, 8,8,2,192, 4,4,1);
    conv(actB, 14, m5, actA, false, 4,4,1,192, 4,4,1,224, 1,1,0);         // w15
    conv(actA, 15, m5, actB, false, 4,4,1,224, 4,4,1,224, 1,1,0);         // w16
    // ---- transposed convs back up, final one writes f32 to d_out ----
    conv(actB, 16, m4, actA, false, 4,4,1,224, 8,8,2,128, 1,0,1);         // w17 deconv
    conv(actA, 17, m3, d_out, true, 8,8,2,128, 16,16,4,64, 1,0,1);        // w18 deconv
}
